// ElectronDensityPredictor_12627203850727
// MI455X (gfx1250) — compile-verified, matrix operand order checked by reading, not run on hardware
//
#include <hip/hip_runtime.h>
#include <math.h>

#define NN 30000
#define EE 480000
#define ET (EE + NN)          // edges + self loops
#define HC 256
#define HEADS 8
#define HID 32
#define EPS 1e-5f
#define NEG_SLOPE 0.2f

typedef __attribute__((ext_vector_type(2)))  float    v2f;
typedef __attribute__((ext_vector_type(8)))  float    v8f;
typedef __attribute__((ext_vector_type(16))) _Float16 v16h;

// ---------------------------------------------------------------- utilities

__global__ void fill_kernel(float* __restrict__ p, float v, int n) {
    int i = blockIdx.x * blockDim.x + threadIdx.x;
    if (i < n) p[i] = v;
}

__device__ __forceinline__ void atomicMaxF(float* addr, float val) {
    // classic sign-magnitude trick; buffer must be initialized to -INF bits
    if (val >= 0.0f) atomicMax((int*)addr, __float_as_int(val));
    else             atomicMin((unsigned int*)addr, __float_as_uint(val));
}

__device__ __forceinline__ float elu1(float x) {
    return x > 0.0f ? x : expm1f(x);
}

// ---------------------------------------------------------------- WMMA GEMM
// C[M x Nc] = A[M x K] @ B[K x Nc], f32. One wave computes one 16x16 tile.
// M = 30000 (divisible by 16), Nc = 256, K in {128, 256}. 8 waves / block.

__global__ void gemm_wmma_f32(const float* __restrict__ A,
                              const float* __restrict__ B,
                              float* __restrict__ C,
                              int K, int Nc) {
    const int lane   = threadIdx.x & 31;
    const int wave   = threadIdx.x >> 5;
    const int tile   = blockIdx.x * 8 + wave;      // grid sized exactly
    const int tilesN = Nc >> 4;
    const int tm     = tile / tilesN;
    const int tn     = tile % tilesN;

    const int mn   = lane & 15;                    // row (A) / col (B,C)
    const int grp  = lane >> 4;                    // half-wave group
    const int m    = tm * 16 + mn;
    const int n    = tn * 16 + mn;

    v8f acc = {};
    const float* arow = A + (size_t)m * K;

#if defined(__HIP_DEVICE_COMPILE__) && __has_builtin(__builtin_amdgcn_wmma_f32_16x16x4_f32)
    const int koff = grp * 2;                      // K offset for this group
    for (int k0 = 0; k0 < K; k0 += 4) {
        v2f a, b;
        // A 16x4: VGPR0 = K=k0+koff, VGPR1 = K=k0+koff+1 (contiguous pair)
        a.x = arow[k0 + koff];
        a.y = arow[k0 + koff + 1];
        // B 4x16: same K split, row-striped across lanes
        b.x = B[(size_t)(k0 + koff)     * Nc + n];
        b.y = B[(size_t)(k0 + koff + 1) * Nc + n];
        acc = __builtin_amdgcn_wmma_f32_16x16x4_f32(
            false, a, false, b, (short)0, acc, false, false);
    }
#elif defined(__HIP_DEVICE_COMPILE__) && __has_builtin(__builtin_amdgcn_wmma_f32_16x16x32_f16)
    // fallback: f16 inputs, f32 accumulate, K step 32
    for (int k0 = 0; k0 < K; k0 += 32) {
        v16h a, b;
        for (int i = 0; i < 16; ++i) {
            int v = i >> 1, j = i & 1;
            int k = (v < 4) ? (2 * v + j + 8 * grp)
                            : (16 + 2 * (v - 4) + j + 8 * grp);
            a[i] = (_Float16)arow[k0 + k];
            b[i] = (_Float16)B[(size_t)(k0 + k) * Nc + n];
        }
        acc = __builtin_amdgcn_wmma_f32_16x16x32_f16(
            false, a, false, b, (short)0, acc, false, false);
    }
#else
    // host compilation pass (never executed) / no-builtin stub:
    // plain scalar accumulate so the TU always compiles.
    for (int k0 = 0; k0 < K; ++k0) {
        float av = arow[k0];
        for (int r = 0; r < 8; ++r)
            acc[r] += av * B[(size_t)k0 * Nc + n];
    }
#endif
    // C/D layout: VGPR r -> row r + 8*grp, col = lane&15
    for (int r = 0; r < 8; ++r) {
        C[(size_t)(tm * 16 + r + 8 * grp) * Nc + n] = acc[r];
    }
}

// ------------------------------------------------- per-node attention logits
// alS[n,h] = sum_c h[n,h,c]*aS[h,c] ; same for alD

__global__ void node_alpha(const float* __restrict__ hbuf,
                           const float* __restrict__ aS,
                           const float* __restrict__ aD,
                           float* __restrict__ alS,
                           float* __restrict__ alD) {
    int idx = blockIdx.x * blockDim.x + threadIdx.x;
    if (idx >= NN * HEADS) return;
    int nd = idx >> 3, h = idx & 7;
    const float* hp = hbuf + (size_t)nd * HC + h * HID;
    const float* as = aS + h * HID;
    const float* ad = aD + h * HID;
    float ss = 0.f, sd = 0.f;
    for (int c = 0; c < HID; ++c) {
        float v = hp[c];
        ss = fmaf(v, as[c], ss);
        sd = fmaf(v, ad[c], sd);
    }
    alS[idx] = ss;
    alD[idx] = sd;
}

// ---------------------------------------------------------------- edge passes

__device__ __forceinline__ void edge_sd(const int* ei, int e, int& s, int& d) {
    if (e < EE) { s = ei[e]; d = ei[EE + e]; }
    else        { s = e - EE; d = s; }           // self loop
}

__global__ void edge_max(const int* __restrict__ ei,
                         const float* __restrict__ alS,
                         const float* __restrict__ alD,
                         float* __restrict__ mx) {
    int idx = blockIdx.x * blockDim.x + threadIdx.x;
    if (idx >= ET * HEADS) return;
    int e = idx >> 3, h = idx & 7;
    int s, d; edge_sd(ei, e, s, d);
    float le = alS[s * HEADS + h] + alD[d * HEADS + h];
    le = le > 0.f ? le : NEG_SLOPE * le;
    atomicMaxF(&mx[d * HEADS + h], le);
}

__global__ void edge_expsum(const int* __restrict__ ei,
                            const float* __restrict__ alS,
                            const float* __restrict__ alD,
                            const float* __restrict__ mx,
                            float* __restrict__ sm,
                            float* __restrict__ ex) {
    int idx = blockIdx.x * blockDim.x + threadIdx.x;
    if (idx >= ET * HEADS) return;
    int e = idx >> 3, h = idx & 7;
    int s, d; edge_sd(ei, e, s, d);
    float le = alS[s * HEADS + h] + alD[d * HEADS + h];
    le = le > 0.f ? le : NEG_SLOPE * le;
    float v = __expf(le - mx[d * HEADS + h]);
    ex[idx] = v;
    atomicAdd(&sm[d * HEADS + h], v);
}

// warp per edge: agg[d, h*32+lane] += h[s, h*32+lane] * alpha(e,h)
__global__ void edge_agg(const int* __restrict__ ei,
                         const float* __restrict__ hbuf,
                         const float* __restrict__ ex,
                         const float* __restrict__ sm,
                         float* __restrict__ agg) {
    int e = blockIdx.x * 8 + (threadIdx.x >> 5);
    int lane = threadIdx.x & 31;
    if (e >= ET) return;
    int s, d; edge_sd(ei, e, s, d);
    const float* hrow = hbuf + (size_t)s * HC;
    float* orow = agg + (size_t)d * HC;
    for (int h = 0; h < HEADS; ++h) {
        float alpha = ex[e * HEADS + h] / sm[d * HEADS + h];
        atomicAdd(&orow[h * HID + lane], hrow[h * HID + lane] * alpha);
    }
}

// ------------------------------------------------------------ bias + ELU, BN

__global__ void bias_elu(float* __restrict__ x, const float* __restrict__ b) {
    int idx = blockIdx.x * blockDim.x + threadIdx.x;
    if (idx >= NN * HC) return;
    x[idx] = elu1(x[idx] + b[idx & (HC - 1)]);
}

__global__ void bn_stats(const float* __restrict__ x,
                         float* __restrict__ s, float* __restrict__ q) {
    int ch = threadIdx.x;                 // blockDim.x == HC
    float a = 0.f, a2 = 0.f;
    for (int r = blockIdx.x; r < NN; r += gridDim.x) {
        float v = x[(size_t)r * HC + ch];
        a += v; a2 = fmaf(v, v, a2);
    }
    atomicAdd(&s[ch], a);
    atomicAdd(&q[ch], a2);
}

__global__ void bn_norm(float* __restrict__ x,
                        const float* __restrict__ s,
                        const float* __restrict__ q,
                        const float* __restrict__ g,
                        const float* __restrict__ be) {
    int idx = blockIdx.x * blockDim.x + threadIdx.x;
    if (idx >= NN * HC) return;
    int ch = idx & (HC - 1);
    float m = s[ch] * (1.0f / NN);
    float v = q[ch] * (1.0f / NN) - m * m;
    x[idx] = (x[idx] - m) * rsqrtf(v + EPS) * g[ch] + be[ch];
}

// ------------------------------------------------------------- MLP head
// warp per node: z = elu(row @ lW1 + lb1) ; out = z @ lW2 + lb2

__global__ void mlp_head(const float* __restrict__ h,
                         const float* __restrict__ lW1,
                         const float* __restrict__ lb1,
                         const float* __restrict__ lW2,
                         const float* __restrict__ lb2,
                         float* __restrict__ out) {
    int node = blockIdx.x * 8 + (threadIdx.x >> 5);
    int lane = threadIdx.x & 31;
    if (node >= NN) return;
    const float* row = h + (size_t)node * HC;
    float z = lb1[lane];
    for (int k = 0; k < HC; ++k)
        z = fmaf(row[k], lW1[k * HID + lane], z);
    z = elu1(z);
    float p = z * lW2[lane];
    for (int off = 16; off; off >>= 1)
        p += __shfl_down(p, off, 32);
    if (lane == 0) out[node] = p + lb2[0];
}

// ---------------------------------------------------------------- launcher

static inline int cdiv(int a, int b) { return (a + b - 1) / b; }

extern "C" void kernel_launch(void* const* d_in, const int* in_sizes, int n_in,
                              void* d_out, int out_size, void* d_ws, size_t ws_size,
                              hipStream_t stream) {
    const float* x   = (const float*)d_in[0];
    const int*   ei  = (const int*)  d_in[1];
    const float* W1  = (const float*)d_in[2];
    const float* a1s = (const float*)d_in[3];
    const float* a1d = (const float*)d_in[4];
    const float* b1  = (const float*)d_in[5];
    const float* g1  = (const float*)d_in[6];
    const float* be1 = (const float*)d_in[7];
    const float* W2  = (const float*)d_in[8];
    const float* a2s = (const float*)d_in[9];
    const float* a2d = (const float*)d_in[10];
    const float* b2  = (const float*)d_in[11];
    const float* g2  = (const float*)d_in[12];
    const float* be2 = (const float*)d_in[13];
    const float* lW1 = (const float*)d_in[14];
    const float* lb1 = (const float*)d_in[15];
    const float* lW2 = (const float*)d_in[16];
    const float* lb2 = (const float*)d_in[17];
    float* out = (float*)d_out;

    float* ws  = (float*)d_ws;
    float* h   = ws;                               // NN*HC
    float* agg = h   + (size_t)NN * HC;            // NN*HC
    float* alS = agg + (size_t)NN * HC;            // NN*HEADS
    float* alD = alS + (size_t)NN * HEADS;         // NN*HEADS
    float* mx  = alD + (size_t)NN * HEADS;         // NN*HEADS
    float* sm  = mx  + (size_t)NN * HEADS;         // NN*HEADS
    float* ex  = sm  + (size_t)NN * HEADS;         // ET*HEADS
    float* bns = ex  + (size_t)ET * HEADS;         // HC
    float* bnq = bns + HC;                         // HC

    const int B = 256;
    const int nEH   = ET * HEADS;
    const int gEH   = cdiv(nEH, B);
    const int gNH   = cdiv(NN * HEADS, B);
    const int gNC   = cdiv(NN * HC, B);
    const int gGEMM = (NN / 16) * (HC / 16) / 8;   // 3750, exact
    const int gEDGE = cdiv(ET, 8);
    const int gNODE = cdiv(NN, 8);

    // ---------------- layer 1 ----------------
    fill_kernel<<<gNH, B, 0, stream>>>(mx, -INFINITY, NN * HEADS);
    fill_kernel<<<gNH, B, 0, stream>>>(sm, 0.f, NN * HEADS);
    fill_kernel<<<gNC, B, 0, stream>>>(agg, 0.f, NN * HC);
    fill_kernel<<<2,   B, 0, stream>>>(bns, 0.f, 2 * HC);

    gemm_wmma_f32<<<gGEMM, B, 0, stream>>>(x, W1, h, 128, HC);
    node_alpha<<<gNH, B, 0, stream>>>(h, a1s, a1d, alS, alD);
    edge_max<<<gEH, B, 0, stream>>>(ei, alS, alD, mx);
    edge_expsum<<<gEH, B, 0, stream>>>(ei, alS, alD, mx, sm, ex);
    edge_agg<<<gEDGE, B, 0, stream>>>(ei, h, ex, sm, agg);
    bias_elu<<<gNC, B, 0, stream>>>(agg, b1);
    bn_stats<<<512, HC, 0, stream>>>(agg, bns, bnq);
    bn_norm<<<gNC, B, 0, stream>>>(agg, bns, bnq, g1, be1);

    // ---------------- layer 2 ----------------
    gemm_wmma_f32<<<gGEMM, B, 0, stream>>>(agg, W2, h, HC, HC);
    node_alpha<<<gNH, B, 0, stream>>>(h, a2s, a2d, alS, alD);
    fill_kernel<<<gNH, B, 0, stream>>>(mx, -INFINITY, NN * HEADS);
    fill_kernel<<<gNH, B, 0, stream>>>(sm, 0.f, NN * HEADS);
    fill_kernel<<<gNC, B, 0, stream>>>(agg, 0.f, NN * HC);
    fill_kernel<<<2,   B, 0, stream>>>(bns, 0.f, 2 * HC);
    edge_max<<<gEH, B, 0, stream>>>(ei, alS, alD, mx);
    edge_expsum<<<gEH, B, 0, stream>>>(ei, alS, alD, mx, sm, ex);
    edge_agg<<<gEDGE, B, 0, stream>>>(ei, h, ex, sm, agg);
    bias_elu<<<gNC, B, 0, stream>>>(agg, b2);
    bn_stats<<<512, HC, 0, stream>>>(agg, bns, bnq);
    bn_norm<<<gNC, B, 0, stream>>>(agg, bns, bnq, g2, be2);

    // ---------------- head ----------------
    mlp_head<<<gNODE, B, 0, stream>>>(agg, lW1, lb1, lW2, lb2, out);
}
